// Encoder_37271726195512
// MI455X (gfx1250) — matrix operand order repaired and checked by reading, not compiled, not run on hardware
//
#include <hip/hip_runtime.h>
#include <hip/hip_bf16.h>

typedef _Float16 v16h __attribute__((ext_vector_type(16)));
typedef _Float16 v8h  __attribute__((ext_vector_type(8)));
typedef float    v8f  __attribute__((ext_vector_type(8)));

#define T_STEPS 64
#define BATCH   16384
#define E_DIM   64
#define H_DIM   128

__device__ __forceinline__ float fast_sigmoid(float x) {
    return __builtin_amdgcn_rcpf(1.0f + __expf(-x));
}
__device__ __forceinline__ float fast_tanh(float x) {
    // tanh(x) = 1 - 2/(exp(2x)+1)
    return 1.0f - 2.0f * __builtin_amdgcn_rcpf(__expf(2.0f * x) + 1.0f);
}

// Gather an A-matrix fragment (16x32 f16, M=row=lane%16) from row-major LDS.
// This code's contiguous k order (same as ld_b_frag): lanes 0-15 hold K {0..15}; lanes 16-31 hold K {16..31}.
__device__ __forceinline__ v16h ld_a_frag(const _Float16* lds, int ldk, int kbase, int lane) {
    int row = lane & 15;
    int sub = (lane & 16) ? 16 : 0;
    const _Float16* p = lds + row * ldk + kbase + sub;
    v8h lo = *(const v8h*)(p);        // K = base+sub .. +7   (16B aligned)
    v8h hi = *(const v8h*)(p + 8);    // K = base+sub+8 .. +15
    v16h r;
#pragma unroll
    for (int i = 0; i < 8; ++i) { r[i] = lo[i]; r[i + 8] = hi[i]; }
    return r;
}

// Build a B-matrix fragment (32x16 f16, N=col=lane%16) from a row-major f32
// weight W (4H x K): B[k][n] = W[n][k].  Lanes 0-15: K 0..15, lanes 16-31: K 16..31,
// VGPR j packs K = {2j, 2j+1}.
__device__ __forceinline__ v16h ld_b_frag(const float* __restrict__ W, int ld,
                                          int ncol, int kbase, int lane) {
    int n  = ncol + (lane & 15);
    int ks = kbase + ((lane & 16) ? 16 : 0);
    const float* p = W + (size_t)n * ld + ks;
    v16h r;
#pragma unroll
    for (int j = 0; j < 8; ++j) {
        float2 f = *(const float2*)(p + 2 * j);
        r[2 * j]     = (_Float16)f.x;
        r[2 * j + 1] = (_Float16)f.y;
    }
    return r;
}

__global__ __launch_bounds__(256, 1)
void lstm_encoder_persistent(const float* __restrict__ obs,
                             const float* __restrict__ Wemb,
                             const float* __restrict__ bemb,
                             const float* __restrict__ wih,
                             const float* __restrict__ whh,
                             const float* __restrict__ bih,
                             const float* __restrict__ bhh,
                             float* __restrict__ out) {
    __shared__ __align__(16) _Float16 hbuf[2][16 * H_DIM];  // h state, f16, double-buffered
    __shared__ __align__(16) _Float16 xbuf[2][16 * E_DIM];  // embedded x_t, f16, double-buffered
    __shared__ float sWe[2 * E_DIM];
    __shared__ float sBe[E_DIM];
    __shared__ float smask[2][16];   // raw obs[...,0] per row (NaN = masked), double-buffered

    const int tid     = threadIdx.x;
    const int lane    = tid & 31;
    const int wave    = tid >> 5;
    const int bbase   = blockIdx.x * 16;   // this WG's 16 batch rows
    const int hidbase = wave * 16;         // this wave's 16 hidden columns

    // Per-thread embedding job: 4 consecutive e's of one row.
    const int erow = tid >> 4;             // (tid*4)>>6
    const int e0   = (tid * 4) & 63;

    // Cache embedding weights in LDS; zero initial h buffer.
    for (int i = tid; i < 2 * E_DIM; i += 256) sWe[i] = Wemb[i];
    for (int i = tid; i < E_DIM; i += 256)     sBe[i] = bemb[i];
    for (int i = tid; i < 16 * H_DIM; i += 256) hbuf[0][i] = (_Float16)0.0f;

    // ---- Persistent weights: convert f32 -> f16 WMMA B-fragments, keep in VGPRs.
    v16h wih_f[4][2];   // 4 gate N-tiles x (E=64 -> 2 K-steps)
    v16h whh_f[4][4];   // 4 gate N-tiles x (H=128 -> 4 K-steps)
    v8f  bias_v[4];
#pragma unroll
    for (int g = 0; g < 4; ++g) {
        const int nb = g * H_DIM + hidbase;
        const float b = bih[nb + (lane & 15)] + bhh[nb + (lane & 15)];
#pragma unroll
        for (int j = 0; j < 8; ++j) bias_v[g][j] = b;
#pragma unroll
        for (int kk = 0; kk < 2; ++kk) wih_f[g][kk] = ld_b_frag(wih, E_DIM, nb, kk * 32, lane);
#pragma unroll
        for (int kk = 0; kk < 4; ++kk) whh_f[g][kk] = ld_b_frag(whh, H_DIM, nb, kk * 32, lane);
    }

    __syncthreads();  // sWe/sBe ready

    // ---- Prologue: embed x_0 into xbuf[0]
    {
        const size_t ob = ((size_t)(bbase + erow)) * 2;  // t = 0
        const float x0 = obs[ob + 0];
        const float x1 = obs[ob + 1];
        if ((tid & 15) == 0) smask[0][erow] = x0;
        const float x0z = (x0 != x0) ? 0.0f : x0;
        const float x1z = (x1 != x1) ? 0.0f : x1;
#pragma unroll
        for (int q = 0; q < 4; ++q) {
            const int e = e0 + q;
            xbuf[0][erow * E_DIM + e] =
                (_Float16)(x0z * sWe[e] + x1z * sWe[E_DIM + e] + sBe[e]);
        }
    }

    // f32 recurrent state, resident in registers for all 64 steps.
    v8f h_reg = {};
    v8f c_reg = {};

    const int rsub = (lane & 16) ? 8 : 0;  // row offset of this lane's C/D slice
    const int ncol = lane & 15;            // column within the 16-wide tile

    __syncthreads();  // xbuf[0] + hbuf[0] ready

    int cur = 0;
    for (int t = 0; t < T_STEPS; ++t) {
        const int par = t & 1;

        // ---- issue obs(t+1) global load NOW; latency hides under the WMMAs below
        const int tn = (t + 1 < T_STEPS) ? (t + 1) : t;
        const size_t obn = ((size_t)tn * BATCH + (size_t)(bbase + erow)) * 2;
        const float x0n = obs[obn + 0];
        const float x1n = obs[obn + 1];

        // ---- A fragments for this step (shared across the 4 gate tiles)
        v16h ax[2], ah[4];
#pragma unroll
        for (int kk = 0; kk < 2; ++kk) ax[kk] = ld_a_frag(xbuf[par], E_DIM, kk * 32, lane);
#pragma unroll
        for (int kk = 0; kk < 4; ++kk) ah[kk] = ld_a_frag(hbuf[cur], H_DIM, kk * 32, lane);

        // ---- gates = x@w_ih^T + h@w_hh^T + b  (24 WMMAs / wave / step)
        v8f acc[4];
#pragma unroll
        for (int g = 0; g < 4; ++g) {
            v8f a = bias_v[g];
#pragma unroll
            for (int kk = 0; kk < 2; ++kk)
                a = __builtin_amdgcn_wmma_f32_16x16x32_f16(false, ax[kk], false, wih_f[g][kk],
                                                           (short)0, a, false, false);
#pragma unroll
            for (int kk = 0; kk < 4; ++kk)
                a = __builtin_amdgcn_wmma_f32_16x16x32_f16(false, ah[kk], false, whh_f[g][kk],
                                                           (short)0, a, false, false);
            acc[g] = a;
        }

        // ---- pipelined embedding for step t+1 -> opposite parity buffers
        {
            if ((tid & 15) == 0) smask[par ^ 1][erow] = x0n;
            const float x0z = (x0n != x0n) ? 0.0f : x0n;
            const float x1z = (x1n != x1n) ? 0.0f : x1n;
#pragma unroll
            for (int q = 0; q < 4; ++q) {
                const int e = e0 + q;
                xbuf[par ^ 1][erow * E_DIM + e] =
                    (_Float16)(x0z * sWe[e] + x1z * sWe[E_DIM + e] + sBe[e]);
            }
        }

        // ---- LSTM cell element-wise + masked state update (f32 in registers)
#pragma unroll
        for (int j = 0; j < 8; ++j) {
            const float iv = fast_sigmoid(acc[0][j]);
            const float fv = fast_sigmoid(acc[1][j]);
            const float gv = fast_tanh(acc[2][j]);
            const float ov = fast_sigmoid(acc[3][j]);
            const float cn = fv * c_reg[j] + iv * gv;
            const float hn = ov * fast_tanh(cn);
            const float m0 = smask[par][j + rsub];
            const bool valid = !(m0 != m0);   // mask = !isnan(obs[t,b,0])
            c_reg[j] = valid ? cn : c_reg[j];
            h_reg[j] = valid ? hn : h_reg[j];
        }

        // ---- publish h (f16) for next step
        const int nxt = cur ^ 1;
#pragma unroll
        for (int j = 0; j < 8; ++j)
            hbuf[nxt][(j + rsub) * H_DIM + hidbase + ncol] = (_Float16)h_reg[j];
        cur = nxt;

        __syncthreads();  // single barrier: hbuf[nxt], xbuf[par^1], smask[par^1] all ready
    }

    // ---- final hidden state, f32, shape (1, B, H)
#pragma unroll
    for (int j = 0; j < 8; ++j)
        out[(size_t)(bbase + j + rsub) * H_DIM + hidbase + ncol] = h_reg[j];
}

extern "C" void kernel_launch(void* const* d_in, const int* in_sizes, int n_in,
                              void* d_out, int out_size, void* d_ws, size_t ws_size,
                              hipStream_t stream) {
    (void)in_sizes; (void)n_in; (void)d_ws; (void)ws_size; (void)out_size;
    const float* obs  = (const float*)d_in[0];  // (T, B, 2)
    const float* Wemb = (const float*)d_in[1];  // (2, E)
    const float* bemb = (const float*)d_in[2];  // (E,)
    const float* wih  = (const float*)d_in[3];  // (4H, E)
    const float* whh  = (const float*)d_in[4];  // (4H, H)
    const float* bih  = (const float*)d_in[5];  // (4H,)
    const float* bhh  = (const float*)d_in[6];  // (4H,)
    float* out = (float*)d_out;                 // (1, B, H)

    dim3 grid(BATCH / 16);   // 1024 workgroups, 16 batch rows each
    dim3 block(256);         // 8 wave32s; wave w owns hidden cols [16w, 16w+16)
    lstm_encoder_persistent<<<grid, block, 0, stream>>>(obs, Wemb, bemb, wih, whh, bih, bhh, out);
}
